// LearnableHarmonicSource_72516227826171
// MI455X (gfx1250) — matrix operand order repaired and checked by reading, not compiled, not run on hardware
//
#include <hip/hip_runtime.h>
#include <math.h>

#define SRATE   48000.0f
#define HOP     480
#define NHARM   6
#define MELCH   128
#define CHUNK   256
#define T_FRM   2000
#define NB      4
#define T_AUD   (T_FRM * HOP)      /* 960000 */
#define NCHUNK  (T_AUD / CHUNK)    /* 3750   */
#define KTOT    645                /* 129 ch * 5 taps */
#define KT_TILES 21                /* ceil(645/32)    */
#define L2E     1.44269504088896f  /* log2(e) */

typedef __attribute__((ext_vector_type(16))) _Float16 v16h;
typedef __attribute__((ext_vector_type(8)))  float    v8f;

#if __has_builtin(__builtin_amdgcn_exp2f)
#define EXP2F(x) __builtin_amdgcn_exp2f(x)
#else
#define EXP2F(x) exp2f(x)
#endif
#define RCPF(x)  __builtin_amdgcn_rcpf(x)
#define SIN2PI(x) __builtin_amdgcn_sinf(x)   /* v_sin_f32: sin(2*pi*x), x in turns */

// consts[] workspace layout
#define C_ESCALE 0            /* exp(amp_logscale[h]), h=0..5 */
#define C_OFFT   NHARM        /* phase offset in turns, h=0..5 */
#define C_NV     (2*NHARM)    /* exp(log_noise_voiced)   */
#define C_NU     (2*NHARM+1)  /* exp(log_noise_unvoiced) */
#define NCONST   (2*NHARM+2)

// linear-interp of per-batch f0 track at audio sample j (scale = 1/480 exact)
__device__ __forceinline__ float f0_at(const float* __restrict__ f0b, int j) {
    float pos = (j + 0.5f) * (1.0f / (float)HOP) - 0.5f;
    pos = fminf(fmaxf(pos, 0.0f), (float)(T_FRM - 1));
    int i0 = (int)floorf(pos);
    int i1 = i0 + 1; if (i1 > T_FRM - 1) i1 = T_FRM - 1;
    float w = pos - (float)i0;
    return f0b[i0] * (1.0f - w) + f0b[i1] * w;
}

// ---------------------------------------------------------------------------
// Kernel 1: conv1 (129x5 -> 32) as WMMA GEMM + silu + conv2 (32 -> 6).
// One wave per (batch, 16-frame tile). M=32 (2 tiles), N=16 frames, K=645.
//   A lane L, elem e: K = e + (L>=16 ? 16 : 0),                M = L&15
//   B lane L, elem e: K = e + (L>=16 ? 16 : 0),                N = L&15
//   C vgpr r, lane L: M = r + (L>=16 ? 8 : 0),                 N = L&15
// ---------------------------------------------------------------------------
__global__ __launch_bounds__(32)
void conv_wmma_kernel(const float* __restrict__ f0, const float* __restrict__ mel,
                      const float* __restrict__ w1, const float* __restrict__ b1,
                      const float* __restrict__ w2, const float* __restrict__ b2,
                      float* __restrict__ amp_mod) {
    const int tile = blockIdx.x;            // 0..124
    const int b    = blockIdx.y;            // 0..3
    const int t0   = tile * 16;
    const int lane = threadIdx.x;           // 0..31
    const int lo   = lane & 15;
    const bool hi  = lane >= 16;

    const float* f0b  = f0  + b * T_FRM;
    const float* melb = mel + b * MELCH * T_FRM;

    v8f acc0 = {};  // out channels 0..15
    v8f acc1 = {};  // out channels 16..31

    for (int kt = 0; kt < KT_TILES; ++kt) {
        v16h a0, a1, bf;
        #pragma unroll
        for (int e = 0; e < 16; ++e) {
            // ---- A fragments (weights, row-major oc*645+kk) ----
            int ka = kt * 32 + e + (hi ? 16 : 0);
            float wa0 = 0.0f, wa1 = 0.0f;
            if (ka < KTOT) {
                wa0 = w1[lo * KTOT + ka];
                wa1 = w1[(lo + 16) * KTOT + ka];
            }
            a0[e] = (_Float16)wa0;
            a1[e] = (_Float16)wa1;
            // ---- B fragment (input patch: kk = c*5 + tap) ----
            int kb = kt * 32 + e + (hi ? 16 : 0);
            float xv = 0.0f;
            if (kb < KTOT) {
                int c  = kb / 5;
                int k  = kb % 5;
                int tt = t0 + lo + k - 2;   // pad=2
                if (tt >= 0 && tt < T_FRM)
                    xv = (c == 0) ? f0b[tt] : melb[(c - 1) * T_FRM + tt];
            }
            bf[e] = (_Float16)xv;
        }
        acc0 = __builtin_amdgcn_wmma_f32_16x16x32_f16(false, a0, false, bf, (short)0, acc0, false, false);
        acc1 = __builtin_amdgcn_wmma_f32_16x16x32_f16(false, a1, false, bf, (short)0, acc1, false, false);
    }

    // silu(h + bias) -> LDS tile sh[oc][frame]  (silu via exp2/rcp pipe)
    __shared__ float sh[32][17];
    #pragma unroll
    for (int r = 0; r < 8; ++r) {
        int oc = r + (hi ? 8 : 0);
        float h0 = acc0[r] + b1[oc];
        float h1 = acc1[r] + b1[oc + 16];
        sh[oc][lo]      = h0 * RCPF(1.0f + EXP2F(-h0 * L2E));
        sh[oc + 16][lo] = h1 * RCPF(1.0f + EXP2F(-h1 * L2E));
    }
    __syncthreads();

    // conv2: 1-tap, 32 -> 6 channels. 16 lanes each own one frame.
    if (lane < 16) {
        int t = t0 + lane;
        #pragma unroll
        for (int j = 0; j < NHARM; ++j) {
            float s = b2[j];
            #pragma unroll
            for (int oc = 0; oc < 32; ++oc)
                s += sh[oc][lane] * w2[j * 32 + oc];
            amp_mod[(b * NHARM + j) * T_FRM + t] = s;
        }
    }
}

// ---------------------------------------------------------------------------
// Kernel 2: per-chunk phase-increment totals (block reduction, 256 thr/chunk)
// ---------------------------------------------------------------------------
__global__ __launch_bounds__(256)
void chunk_totals_kernel(const float* __restrict__ f0, float* __restrict__ totals) {
    const int chunk = blockIdx.x;
    const int b     = blockIdx.y;
    const int j     = chunk * CHUNK + threadIdx.x;
    const float* f0b = f0 + b * T_FRM;

    float p = f0_at(f0b, j) * (1.0f / SRATE);
    #pragma unroll
    for (int off = 16; off > 0; off >>= 1)
        p += __shfl_down(p, off, 32);

    __shared__ float wsum[8];
    int wid = threadIdx.x >> 5, lane = threadIdx.x & 31;
    if (lane == 0) wsum[wid] = p;
    __syncthreads();
    if (threadIdx.x == 0) {
        float s = 0.0f;
        #pragma unroll
        for (int w = 0; w < 8; ++w) s += wsum[w];
        totals[b * NCHUNK + chunk] = s;
    }
}

// ---------------------------------------------------------------------------
// Kernel 3a: tiny per-harmonic constants precompute (1 wave)
// ---------------------------------------------------------------------------
__global__ void consts_kernel(const float* __restrict__ als, const float* __restrict__ poff,
                              const float* __restrict__ lnv, const float* __restrict__ lnu,
                              float* __restrict__ consts) {
    int t = threadIdx.x;
    if (t < NHARM) {
        consts[C_ESCALE + t] = EXP2F(als[t] * L2E);   // exp(amp_logscale[h])
        consts[C_OFFT + t]   = poff[t];               // offset already in turns
    } else if (t == NHARM) {
        consts[C_NV] = EXP2F(lnv[0] * L2E);
    } else if (t == NHARM + 1) {
        consts[C_NU] = EXP2F(lnu[0] * L2E);
    }
}

// ---------------------------------------------------------------------------
// Kernel 3b: PARALLEL carry computation. The reference's sequential
//   carry_{i+1} = fmod(carry_i + total_i, 1)   (all terms >= 0)
// equals carry_i = fract(prefix_exclusive_i) in exact arithmetic, so we use a
// hierarchical block scan (4 chunks/thread -> wave32 shuffle scan -> wave-total
// scan by wave 0) instead of a 3750-step serial dependence chain (~30us of pure
// latency on one wave -- 6x the whole problem's memory roofline).
// One 1024-thread block per batch.
// ---------------------------------------------------------------------------
__global__ __launch_bounds__(1024)
void carry_scan_kernel(const float* __restrict__ totals, float* __restrict__ carries) {
    const int b    = blockIdx.x;
    const int tid  = threadIdx.x;
    const int lane = tid & 31, wid = tid >> 5;
    const float* tb = totals + b * NCHUNK;
    const int c0 = tid * 4;                      // 1024*4 = 4096 >= 3750

    float l[4];
    #pragma unroll
    for (int i = 0; i < 4; ++i)
        l[i] = (c0 + i < NCHUNK) ? tb[c0 + i] : 0.0f;
    float s = (l[0] + l[1]) + (l[2] + l[3]);

    // wave32 inclusive scan of per-thread sums
    float incl = s;
    #pragma unroll
    for (int off = 1; off < 32; off <<= 1) {
        float n = __shfl_up(incl, off, 32);
        if (lane >= off) incl += n;
    }
    __shared__ float wtot[32];
    if (lane == 31) wtot[wid] = incl;
    __syncthreads();
    if (tid < 32) {                              // wave 0 scans the 32 wave totals
        float v = wtot[tid];
        #pragma unroll
        for (int off = 1; off < 32; off <<= 1) {
            float n = __shfl_up(v, off, 32);
            if (lane >= off) v += n;
        }
        wtot[tid] = v;                           // inclusive wave-total prefix
    }
    __syncthreads();

    float run = incl - s + ((wid > 0) ? wtot[wid - 1] : 0.0f);  // exclusive prefix
    #pragma unroll
    for (int i = 0; i < 4; ++i) {
        if (c0 + i < NCHUNK)
            carries[b * NCHUNK + c0 + i] = run - floorf(run);   // fract = carry BEFORE chunk
        run += l[i];
    }
}

// ---------------------------------------------------------------------------
// Kernel 4 (bandwidth-bound): fused phase scan + harmonic synthesis + noise.
// A 256-sample chunk spans <= 3 frames of the 480x-upsampled grid, so f0 and
// amp_mod frames are staged to LDS once per block. All phases kept in TURNS;
// v_sin_f32 computes sin(2*pi*x) directly -> no range reduction anywhere.
// ---------------------------------------------------------------------------
__global__ __launch_bounds__(256)
void synth_kernel(const float* __restrict__ f0, const float* __restrict__ noise,
                  const float* __restrict__ amp_mod, const float* __restrict__ carries,
                  const float* __restrict__ consts, float* __restrict__ out) {
    const int chunk = blockIdx.x;
    const int b     = blockIdx.y;
    const int tid   = threadIdx.x;
    const int j     = chunk * CHUNK + tid;

    __shared__ float s_amp[NHARM][3];
    __shared__ float s_f0[3];
    __shared__ float s_c[NCONST + 1];    // consts + carry
    __shared__ float wsum[8];

    // block-uniform base frame for this chunk
    float posb = (float)(chunk * CHUNK) * (1.0f / (float)HOP) + (0.5f / (float)HOP) - 0.5f;
    posb = fminf(fmaxf(posb, 0.0f), (float)(T_FRM - 1));
    const int base = (int)floorf(posb);

    if (tid < NHARM * 3) {                       // 0..17: amp_mod frames
        int h = tid / 3, r = tid % 3;
        int fr = base + r; if (fr > T_FRM - 1) fr = T_FRM - 1;
        s_amp[h][r] = amp_mod[(b * NHARM + h) * T_FRM + fr];
    } else if (tid < NHARM * 3 + 3) {            // 18..20: f0 frames
        int r = tid - NHARM * 3;
        int fr = base + r; if (fr > T_FRM - 1) fr = T_FRM - 1;
        s_f0[r] = f0[b * T_FRM + fr];
    } else if (tid == 31) {                      // per-chunk carry
        s_c[NCONST] = carries[b * NCHUNK + chunk];
    } else if (tid >= 32 && tid < 32 + NCONST) { // second wave: constants
        s_c[tid - 32] = consts[tid - 32];
    }
    __builtin_prefetch(noise + (size_t)b * T_AUD + j, 0, 3);
    __syncthreads();

    // interp position (identical grid for f0 and amp_mod)
    float pos = (float)j * (1.0f / (float)HOP) + (0.5f / (float)HOP) - 0.5f;
    pos = fminf(fmaxf(pos, 0.0f), (float)(T_FRM - 1));
    int i0 = (int)floorf(pos);
    float w = pos - (float)i0;
    int i1 = i0 + 1; if (i1 > T_FRM - 1) i1 = T_FRM - 1;
    const int r0 = i0 - base;                    // 0..1
    const int r1 = i1 - base;                    // 0..2

    float fA  = s_f0[r0];
    float f0u = fA + w * (s_f0[r1] - fA);
    float v   = f0u * (1.0f / SRATE);

    // inclusive scan over the chunk (wave32 scan + wave prefix)
    int lane = tid & 31, wid = tid >> 5;
    #pragma unroll
    for (int off = 1; off < 32; off <<= 1) {
        float n = __shfl_up(v, off, 32);
        if (lane >= off) v += n;
    }
    if (lane == 31) wsum[wid] = v;
    __syncthreads();
    float pre = 0.0f;
    #pragma unroll
    for (int ww = 0; ww < 8; ++ww)
        if (ww < wid) pre += wsum[ww];
    v += pre;

    float x  = v + s_c[NCONST];
    float pt = x - floorf(x);                    // base phase in TURNS

    const float voiced = (f0u > 1.0f) ? 1.0f : 0.0f;
    // aa = sigmoid((nyq - k*f0u)/(nyq*0.05)); exp2 form with folded constants
    const float aa_k = L2E / (SRATE * 0.5f * 0.05f);   // per-unit-f0u slope
    const float aa_c = (SRATE * 0.5f) * aa_k;          // nyq * slope

    #pragma unroll
    for (int h = 0; h < NHARM; ++h) {
        float kk = (float)(h + 1);
        float a0 = s_amp[h][r0];
        float a  = a0 + w * (s_amp[h][r1] - a0);
        float sig2 = 2.0f * RCPF(1.0f + EXP2F(-a * L2E));            // 2*sigmoid(amp)
        float aa   = RCPF(1.0f + EXP2F(f0u * (kk * aa_k) - aa_c));   // anti-alias mask
        float s    = SIN2PI(kk * pt + s_c[C_OFFT + h]);              // sin(2pi(k*pt+off))
        float val  = s * (s_c[C_ESCALE + h] * voiced) * sig2 * aa;
        out[((size_t)b * (NHARM + 1) + h) * T_AUD + j] = val;
    }
    float namp = (voiced != 0.0f) ? s_c[C_NV] : s_c[C_NU];
    out[((size_t)b * (NHARM + 1) + NHARM) * T_AUD + j] = noise[(size_t)b * T_AUD + j] * namp;
}

// ---------------------------------------------------------------------------
extern "C" void kernel_launch(void* const* d_in, const int* in_sizes, int n_in,
                              void* d_out, int out_size, void* d_ws, size_t ws_size,
                              hipStream_t stream) {
    const float* f0    = (const float*)d_in[0];
    const float* mel   = (const float*)d_in[1];
    const float* noise = (const float*)d_in[2];
    const float* w1    = (const float*)d_in[3];
    const float* b1    = (const float*)d_in[4];
    const float* w2    = (const float*)d_in[5];
    const float* b2    = (const float*)d_in[6];
    const float* als   = (const float*)d_in[7];
    const float* poff  = (const float*)d_in[8];
    const float* lnv   = (const float*)d_in[9];
    const float* lnu   = (const float*)d_in[10];
    float* out = (float*)d_out;

    float* ws      = (float*)d_ws;
    float* amp_mod = ws;                                 // NB*NHARM*T_FRM = 48000 floats
    float* totals  = amp_mod + NB * NHARM * T_FRM;       // NB*NCHUNK = 15000 floats
    float* carries = totals  + NB * NCHUNK;              // NB*NCHUNK = 15000 floats
    float* consts  = carries + NB * NCHUNK;              // NCONST = 14 floats

    conv_wmma_kernel<<<dim3(T_FRM / 16, NB), 32, 0, stream>>>(f0, mel, w1, b1, w2, b2, amp_mod);
    chunk_totals_kernel<<<dim3(NCHUNK, NB), CHUNK, 0, stream>>>(f0, totals);
    consts_kernel<<<1, 32, 0, stream>>>(als, poff, lnv, lnu, consts);
    carry_scan_kernel<<<NB, 1024, 0, stream>>>(totals, carries);
    synth_kernel<<<dim3(NCHUNK, NB), CHUNK, 0, stream>>>(f0, noise, amp_mod, carries, consts, out);
}
